// ConnectedConv_16887811408678
// MI455X (gfx1250) — matrix operand order repaired and checked by reading, not compiled, not run on hardware
//
#include <hip/hip_runtime.h>

// ---------------- static problem config (matches reference) ----------------
#define IN_C    128
#define OUT_C   128
#define POSENC  10
#define L_LEN   8192
#define BATCH   8
#define KWIN    3
#define BASE_C  266               // 2*IN_C + POSENC
#define KPAD    288               // BASE_C padded to multiple of 32 (9 k-tiles)
#define CH_IN   798               // BASE_C * KWIN (W inner dim, index = c*3 + k)
#define TILE_N  128               // L positions per workgroup
#define NCOL    (TILE_N + 2)      // base columns l0-1 .. l0+TILE_N
#define COLH    296               // LDS column stride in halves (592 B, 16B aligned)

typedef __bf16        v16bf __attribute__((ext_vector_type(16)));
typedef float         v8f   __attribute__((ext_vector_type(8)));
typedef unsigned int  v4u   __attribute__((ext_vector_type(4)));

union Frag { v4u u[2]; v16bf v; };   // 32 bytes = 8 VGPRs per lane

__device__ __forceinline__ unsigned short f2bf(float f) {
    unsigned u = __builtin_bit_cast(unsigned, f);
    u += 0x7FFFu + ((u >> 16) & 1u);          // round-to-nearest-even
    return (unsigned short)(u >> 16);
}

// ------------- prep: W[o, c*3+k] -> Wb[k][o][c] bf16, K padded -------------
__global__ __launch_bounds__(256)
void cc_prep_w(const float* __restrict__ W, unsigned short* __restrict__ Wb) {
    int idx = blockIdx.x * 256 + threadIdx.x;
    if (idx >= KWIN * OUT_C * KPAD) return;
    int k   = idx / (OUT_C * KPAD);
    int rem = idx % (OUT_C * KPAD);
    int o   = rem / KPAD;
    int c   = rem % KPAD;
    float v = (c < BASE_C) ? W[o * CH_IN + c * KWIN + k] : 0.0f;
    Wb[idx] = f2bf(v);
}

// ---------------------------- fused main kernel ----------------------------
// grid = (L/TILE_N, B), block = 256 (8 waves).
// Wave tiling: waves arranged 4 (N) x 2 (M); each wave owns a 32x64 (NxM)
// subtile = 2 N-tiles x 4 M-tiles -> af reused x2, bf reused x4.
__global__ __launch_bounds__(256)
void cc_gemm(const float* __restrict__ in, const int* __restrict__ conn,
             const unsigned char* __restrict__ mask,
             const unsigned short* __restrict__ Wb,
             const float* __restrict__ bias, float* __restrict__ out) {
    __shared__ unsigned short s_base[NCOL * COLH];   // bf16, column-major
    __shared__ float s_bias[OUT_C];
    __shared__ float s_mask[TILE_N];
    __shared__ int   s_conn[NCOL];

    const int tid = threadIdx.x;
    const int l0  = blockIdx.x * TILE_N;
    const int b   = blockIdx.y;

    // ---- stage bias / mask / connection indices ----
    if (tid < OUT_C)  s_bias[tid] = bias[tid];
    if (tid < TILE_N) s_mask[tid] = mask[b * L_LEN + l0 + tid] ? 1.0f : 0.0f;
    if (tid < NCOL) {
        int pos = l0 - 1 + tid;
        s_conn[tid] = (pos >= 0 && pos < L_LEN) ? conn[b * L_LEN + pos] : 0;
    }
    __syncthreads();

    // ---- phase 1: build base feature matrix (288 x 130) in LDS, bf16 ----
    // rows 0..127 inputs | 128..255 gathered | 256..265 sin posenc | 266..287 zero
    const float* inB = in + (size_t)b * IN_C * L_LEN;
    for (int idx = tid; idx < KPAD * NCOL; idx += 256) {
        int col = idx % NCOL;          // consecutive tids -> consecutive L: coalesced
        int row = idx / NCOL;
        int pos = l0 - 1 + col;
        float v = 0.0f;
        if (pos >= 0 && pos < L_LEN) {
            if (row < IN_C) {
                v = inB[(size_t)row * L_LEN + pos];
            } else if (row < 2 * IN_C) {
                v = inB[(size_t)(row - IN_C) * L_LEN + s_conn[col]];
            } else if (row < BASE_C) {
                int p = row - 2 * IN_C;
                float delta = (float)pos - (float)s_conn[col];
                v = __sinf((float)(1 << p) * delta * 0.001f);
            }
        }
        s_base[col * COLH + row] = f2bf(v);
    }
    __syncthreads();

    // ---- phase 2: GEMM, K = 3 windows x 288 ----
    const int lane  = tid & 31;
    const int wave  = tid >> 5;
    const int ngrp  = wave & 3;        // 4 N-groups of 32 columns
    const int mgrp  = wave >> 2;       // 2 M-groups of 64 rows
    const int nlane = lane & 15;       // N (B frag) / M (A frag) index within tile
    const int khalf = lane >> 4;

    v8f acc[4][2] = {};

    #pragma unroll
    for (int kw = 0; kw < KWIN; ++kw) {
        const unsigned short* Wk = Wb + (size_t)kw * OUT_C * KPAD;
        // conv tap kw at output position l reads base column (l - l0) + kw
        const unsigned short* bcol0 = s_base + (ngrp * 32 + nlane + kw) * COLH;
        const unsigned short* bcol1 = bcol0 + 16 * COLH;
        for (int kt = 0; kt < KPAD / 32; ++kt) {
            const int koff = kt * 32;
            // B fragments: K rows koff + khalf*16 .. +15, one column per lane
            Frag bf[2];
            {
                const v4u* p0 = (const v4u*)(bcol0 + koff + khalf * 16);
                const v4u* p1 = (const v4u*)(bcol1 + koff + khalf * 16);
                bf[0].u[0] = p0[0]; bf[0].u[1] = p0[1];
                bf[1].u[0] = p1[0]; bf[1].u[1] = p1[1];
            }
            // A fragments: distinct register sets -> one load clause,
            // partial s_wait_loadcnt before each WMMA (latency overlap)
            Frag af[4];
            #pragma unroll
            for (int mt = 0; mt < 4; ++mt) {
                const unsigned short* ap =
                    Wk + (size_t)(mgrp * 64 + mt * 16 + nlane) * KPAD
                       + koff + khalf * 16;
                af[mt].u[0] = *(const v4u*)ap;
                af[mt].u[1] = *(const v4u*)(ap + 8);
            }
            #pragma unroll
            for (int mt = 0; mt < 4; ++mt)
                #pragma unroll
                for (int nt = 0; nt < 2; ++nt)
                    acc[mt][nt] = __builtin_amdgcn_wmma_f32_16x16x32_bf16(
                        false, af[mt].v, false, bf[nt].v,
                        (short)0, acc[mt][nt], false, false);
        }
    }

    // ---- phase 3: bias + mask epilogue, coalesced stores ----
    #pragma unroll
    for (int nt = 0; nt < 2; ++nt) {
        const int N = ngrp * 32 + nt * 16 + nlane;
        const float mk = s_mask[N];
        float* outp = out + (size_t)b * OUT_C * L_LEN + (l0 + N);
        #pragma unroll
        for (int mt = 0; mt < 4; ++mt)
            #pragma unroll
            for (int r = 0; r < 8; ++r) {
                int M = mgrp * 64 + mt * 16 + khalf * 8 + r;  // ISA 7.12.2 C/D layout
                outp[(size_t)M * L_LEN] = (acc[mt][nt][r] + s_bias[M]) * mk;
            }
    }
}

// ------------------------------- launcher ---------------------------------
extern "C" void kernel_launch(void* const* d_in, const int* in_sizes, int n_in,
                              void* d_out, int out_size, void* d_ws, size_t ws_size,
                              hipStream_t stream) {
    const float*         in   = (const float*)d_in[0];
    const int*           conn = (const int*)d_in[1];
    const unsigned char* mask = (const unsigned char*)d_in[2];  // jnp bool -> 1 byte
    const float*         W    = (const float*)d_in[3];
    const float*         bias = (const float*)d_in[4];
    float*               out  = (float*)d_out;
    unsigned short*      Wb   = (unsigned short*)d_ws;          // 3*128*288*2 = 221 KB

    const int prep_elems = KWIN * OUT_C * KPAD;
    cc_prep_w<<<(prep_elems + 255) / 256, 256, 0, stream>>>(W, Wb);

    dim3 grid(L_LEN / TILE_N, BATCH);
    cc_gemm<<<grid, 256, 0, stream>>>(in, conn, mask, Wb, bias, out);
}
